// BitLinear158_46213848106050
// MI455X (gfx1250) — 8-bit matrix operand order repaired and checked by reading, compiled, not run on hardware
//
#include <hip/hip_runtime.h>
#include <stdint.h>

#define DIN   2048
#define DOUT  2048
#define NTOK  8192
#define EPSQ  1e-5f
#define QF    128.0f

#define BLK_M 256
#define BLK_N 64
#define KTILES (DIN / 64)
#define LDSB_STRIDE 80
#define LDSB_BUF   (64 * LDSB_STRIDE)

typedef int v8i __attribute__((ext_vector_type(8)));
typedef int v4i __attribute__((ext_vector_type(4)));

#if defined(__HIP_DEVICE_COMPILE__)
#  if __has_builtin(__builtin_amdgcn_global_load_async_to_lds_b128)
#    define USE_ASYNC_COPY 1
#  endif
#endif

__device__ __forceinline__ void async_copy_b128(const char* g, char* l) {
#if defined(USE_ASYNC_COPY)
    __builtin_amdgcn_global_load_async_to_lds_b128((v4i*)g, (v4i*)l, 0, 0);
#else
    *(int4*)l = *(const int4*)g;
#endif
}

__device__ __forceinline__ void wait_async0() {
#if defined(USE_ASYNC_COPY)
#  if __has_builtin(__builtin_amdgcn_s_wait_asynccnt)
    __builtin_amdgcn_s_wait_asynccnt(0);
#  else
    asm volatile("s_wait_asynccnt 0x0" ::: "memory");
#  endif
#endif
}

__device__ __forceinline__ int pack4(float a, float b, float c, float d) {
    return  ((int)(char)(int)a & 0xff)        |
           (((int)(char)(int)b & 0xff) << 8)  |
           (((int)(char)(int)c & 0xff) << 16) |
           (((int)(char)(int)d & 0xff) << 24);
}

__global__ __launch_bounds__(256) void absum_kernel(const float4* __restrict__ w4,
                                                    float* __restrict__ sum, int n4) {
    __shared__ float red[256];
    float s = 0.f;
    for (int i = blockIdx.x * blockDim.x + threadIdx.x; i < n4; i += gridDim.x * blockDim.x) {
        float4 v = w4[i];
        s += fabsf(v.x) + fabsf(v.y) + fabsf(v.z) + fabsf(v.w);
    }
    red[threadIdx.x] = s;
    __syncthreads();
    for (int off = 128; off > 0; off >>= 1) {
        if ((int)threadIdx.x < off) red[threadIdx.x] += red[threadIdx.x + off];
        __syncthreads();
    }
    if (threadIdx.x == 0) atomicAdd(sum, red[0]);
}

__global__ __launch_bounds__(256) void quantw_kernel(const float4* __restrict__ w4,
                                                     const float* __restrict__ sum,
                                                     int* __restrict__ qw4, int n4) {
    const float scale = fmaxf(sum[0] * (1.0f / (float)(DIN * DOUT)), EPSQ);
    const float inv = 1.0f / scale;
    for (int i = blockIdx.x * blockDim.x + threadIdx.x; i < n4; i += gridDim.x * blockDim.x) {
        float4 v = w4[i];
        float a = fminf(fmaxf(rintf(v.x * inv), -1.f), 1.f);
        float b = fminf(fmaxf(rintf(v.y * inv), -1.f), 1.f);
        float c = fminf(fmaxf(rintf(v.z * inv), -1.f), 1.f);
        float d = fminf(fmaxf(rintf(v.w * inv), -1.f), 1.f);
        qw4[i] = pack4(a, b, c, d);
    }
}

__global__ __launch_bounds__(256) void quantx_kernel(const float* __restrict__ x,
                                                     char* __restrict__ qx,
                                                     float* __restrict__ gamma) {
    __shared__ float red[256];
    const int tok = blockIdx.x;
    const float4* xr4 = (const float4*)(x + (size_t)tok * DIN);
    float m = 0.f;
    float4 v0 = xr4[threadIdx.x];
    float4 v1 = xr4[threadIdx.x + 256];
    m = fmaxf(m, fmaxf(fmaxf(fabsf(v0.x), fabsf(v0.y)), fmaxf(fabsf(v0.z), fabsf(v0.w))));
    m = fmaxf(m, fmaxf(fmaxf(fabsf(v1.x), fabsf(v1.y)), fmaxf(fabsf(v1.z), fabsf(v1.w))));
    red[threadIdx.x] = m;
    __syncthreads();
    for (int off = 128; off > 0; off >>= 1) {
        if ((int)threadIdx.x < off) red[threadIdx.x] = fmaxf(red[threadIdx.x], red[threadIdx.x + off]);
        __syncthreads();
    }
    const float g = fmaxf(red[0], EPSQ);
    if (threadIdx.x == 0) gamma[tok] = g;
    const float s = QF / g;
    int* qr4 = (int*)(qx + (size_t)tok * DIN);
    {
        float a = fminf(fmaxf(rintf(v0.x * s), -QF), QF - 1.f);
        float b = fminf(fmaxf(rintf(v0.y * s), -QF), QF - 1.f);
        float c = fminf(fmaxf(rintf(v0.z * s), -QF), QF - 1.f);
        float d = fminf(fmaxf(rintf(v0.w * s), -QF), QF - 1.f);
        qr4[threadIdx.x] = pack4(a, b, c, d);
    }
    {
        float a = fminf(fmaxf(rintf(v1.x * s), -QF), QF - 1.f);
        float b = fminf(fmaxf(rintf(v1.y * s), -QF), QF - 1.f);
        float c = fminf(fmaxf(rintf(v1.z * s), -QF), QF - 1.f);
        float d = fminf(fmaxf(rintf(v1.w * s), -QF), QF - 1.f);
        qr4[threadIdx.x + 256] = pack4(a, b, c, d);
    }
}

__device__ __forceinline__ void load_afrag(v8i& a, const char* ap) {
    int4 t0 = *(const int4*)(ap);
    int4 t1 = *(const int4*)(ap + 32);
    a[0] = t0.x; a[1] = t0.y; a[2] = t0.z; a[3] = t0.w;
    a[4] = t1.x; a[5] = t1.y; a[6] = t1.z; a[7] = t1.w;
}

__global__ __launch_bounds__(256) void gemm_kernel(const char* __restrict__ qx,
                                                   const char* __restrict__ qw,
                                                   const float* __restrict__ gamma,
                                                   const float* __restrict__ sumw,
                                                   const float* __restrict__ bias,
                                                   float* __restrict__ out) {
    __shared__ char ldsB[2 * LDSB_BUF];

    const int tid   = threadIdx.x;
    const int lane  = tid & 31;
    const int wave  = tid >> 5;
    const int mBase = blockIdx.x * BLK_M + wave * 32;
    const int nBase = blockIdx.y * BLK_N;
    const int hi    = lane >> 4;
    const int lo    = lane & 15;

    const int cn  = tid >> 2;
    const int ckp = (tid & 3) * 16;
    const char* gB = qw + (size_t)(nBase + cn) * DIN + ckp;
    char* lB = &ldsB[cn * LDSB_STRIDE + ckp];

    v8i acc0[4], acc1[4];
#pragma unroll
    for (int j = 0; j < 4; ++j)
#pragma unroll
        for (int r = 0; r < 8; ++r) { acc0[j][r] = 0; acc1[j][r] = 0; }

    const char* arow0 = qx + (size_t)(mBase + lo) * DIN + hi * 16;
    const char* arow1 = arow0 + (size_t)16 * DIN;

    v8i aC[2], aN[2];

    async_copy_b128(gB, lB);
    load_afrag(aC[0], arow0);
    load_afrag(aC[1], arow1);
    wait_async0();
    __syncthreads();

#pragma unroll 1
    for (int t = 0; t < KTILES; ++t) {
        const int kt  = t * 64;
        const int cur = t & 1;
        const int nxt = cur ^ 1;

        if (t + 1 < KTILES) {
            async_copy_b128(gB + kt + 64, lB + nxt * LDSB_BUF);
            load_afrag(aN[0], arow0 + kt + 64);
            load_afrag(aN[1], arow1 + kt + 64);
        }

        const char* bufC = &ldsB[cur * LDSB_BUF];
#pragma unroll
        for (int j = 0; j < 4; ++j) {
            const char* bp = bufC + (j * 16 + lo) * LDSB_STRIDE + hi * 16;
            int4 blo = *(const int4*)(bp);
            int4 bhi = *(const int4*)(bp + 32);
            v8i b;
            b[0] = blo.x; b[1] = blo.y; b[2] = blo.z; b[3] = blo.w;
            b[4] = bhi.x; b[5] = bhi.y; b[6] = bhi.z; b[7] = bhi.w;
            acc0[j] = __builtin_amdgcn_wmma_i32_16x16x64_iu8(true, aC[0], true, b, acc0[j], false, false);
            acc1[j] = __builtin_amdgcn_wmma_i32_16x16x64_iu8(true, aC[1], true, b, acc1[j], false, false);
        }

        wait_async0();
        __syncthreads();

        aC[0] = aN[0];
        aC[1] = aN[1];
    }

    const float scale = fmaxf(sumw[0] * (1.0f / (float)(DIN * DOUT)), EPSQ);
    float g0[8], g1[8];
#pragma unroll
    for (int r = 0; r < 8; ++r) {
        g0[r] = gamma[mBase + r + 8 * hi] * scale * (1.0f / QF);
        g1[r] = gamma[mBase + 16 + r + 8 * hi] * scale * (1.0f / QF);
    }

#pragma unroll
    for (int j = 0; j < 4; ++j) {
        const int n = nBase + j * 16 + lo;
        const float bv = bias[n];
#pragma unroll
        for (int r = 0; r < 8; ++r) {
            const int m0 = mBase + r + 8 * hi;
            out[(size_t)m0 * DOUT + n] = ((float)acc0[j][r] + bv) * g0[r];
            out[(size_t)(m0 + 16) * DOUT + n] = ((float)acc1[j][r] + bv) * g1[r];
        }
    }
}

extern "C" void kernel_launch(void* const* d_in, const int* in_sizes, int n_in,
                              void* d_out, int out_size, void* d_ws, size_t ws_size,
                              hipStream_t stream) {
    const float* x    = (const float*)d_in[0];
    const float* w    = (const float*)d_in[1];
    const float* bias = (const float*)d_in[2];
    float* out = (float*)d_out;

    char* ws = (char*)d_ws;
    float* sum   = (float*)ws;
    float* gamma = (float*)(ws + 256);
    char*  qw    = ws + 65536;
    char*  qx    = ws + 65536 + 4 * 1024 * 1024;

    (void)hipMemsetAsync(sum, 0, sizeof(float), stream);
    absum_kernel<<<1024, 256, 0, stream>>>((const float4*)w, sum, (DIN * DOUT) / 4);
    quantw_kernel<<<1024, 256, 0, stream>>>((const float4*)w, sum, (int*)qw, (DIN * DOUT) / 4);
    quantx_kernel<<<NTOK, 256, 0, stream>>>(x, qx, gamma);

    dim3 grid(NTOK / BLK_M, DOUT / BLK_N);
    gemm_kernel<<<grid, dim3(256), 0, stream>>>(qx, qw, gamma, sum, bias, out);
}
